// DynamicPrunedViT_62646392979714
// MI455X (gfx1250) — compile-verified, matrix operand order checked by reading, not run on hardware
//
#include <hip/hip_runtime.h>
#include <hip/hip_bf16.h>
#include <math.h>

// ---------------- problem constants ----------------
#define BATCH   64
#define NPATCH  196      // 14x14
#define S1      197      // tokens before pruning (cls + 196)
#define S2      99       // tokens after pruning  (cls + 98)
#define KEEPN   98
#define DMODEL  768
#define H_HEADS 12
#define HDIM    64
#define FFDIM   3072
#define NCLS    1000
#define NCLS_P  1024     // padded to multiple of 256 for the head GEMM
#define NLAYER  12
#define PRUNE_L 6

typedef __attribute__((ext_vector_type(16))) __bf16 vbf16_t;
typedef __attribute__((ext_vector_type(8)))  float  vf8_t;

union FragBF { unsigned int u[8]; vbf16_t v; };

__device__ __forceinline__ unsigned short f32_to_bf16(float f) {
    union { float f; unsigned int u; } v; v.f = f;
    unsigned int r = v.u + 0x7FFFu + ((v.u >> 16) & 1u);   // RNE
    return (unsigned short)(r >> 16);
}

// CDNA5 async copy: 16 bytes global -> LDS, tracked by ASYNCcnt.
__device__ __forceinline__ void async_b128(unsigned lds_off, const void* gaddr) {
    asm volatile("global_load_async_to_lds_b128 %0, %1, off"
                 :: "v"(lds_off), "v"(gaddr) : "memory");
}
__device__ __forceinline__ void wait_async0() {
    asm volatile("s_wait_asynccnt 0x0" ::: "memory");
}

// =====================================================================
// One-shot weight convert + transpose:  W f32[K,N] -> Wt bf16[Np,K]
// (rows n >= N zero-filled). Grid = (Np/32, K/32), block 256.
// =====================================================================
__global__ __launch_bounds__(256)
void wconv_kernel(const float* __restrict__ W, unsigned short* __restrict__ Wt,
                  int K, int N)
{
    __shared__ float tile[32][33];
    const int tn = blockIdx.x * 32;
    const int tk = blockIdx.y * 32;
    const int tid = threadIdx.x;
    #pragma unroll
    for (int i = 0; i < 4; ++i) {
        int idx = tid + i * 256;
        int kk = idx >> 5, nn = idx & 31;
        int gn = tn + nn;
        tile[kk][nn] = (gn < N) ? W[(size_t)(tk + kk) * N + gn] : 0.0f;
    }
    __syncthreads();
    #pragma unroll
    for (int i = 0; i < 4; ++i) {
        int idx = tid + i * 256;
        int nn = idx >> 5, kk = idx & 31;
        Wt[(size_t)(tn + nn) * K + tk + kk] = f32_to_bf16(tile[kk][nn]);
    }
}

// =====================================================================
// Tiled bf16 WMMA GEMM:  out = epi( A[M,K](bf16) @ Wt[N,K](bf16, transposed) )
// Block tile 64(M) x 256(N), K-stage 64 (2 WMMA k-steps). 8 waves, each owns
// a 32x64 slab -> 8 v_wmma per k-step. Software-pipelined double-buffered
// staging via global_load_async_to_lds_b128 (copies for stage st+1 overlap
// the WMMA work of stage st); LDS row stride 144B (bank-conflict-free pairs,
// 16B-aligned async chunks).
// EPI: 0 = f32 out + bias                (qkv / head)
//      1 = patch-embed: bias + pos-embed, scatter into t[b,1+tok,:]
//      2 = f32 out = resid + acc + bias  (proj / fc2 residual)
//      3 = bf16 out = gelu(acc + bias)   (fc1)
// =====================================================================
__device__ __forceinline__ void stage_copy(unsigned ldsA, unsigned ldsB,
                                           const unsigned short* __restrict__ A, int lda,
                                           const unsigned short* __restrict__ Wt, int K,
                                           int m0, int n0, int k0, int tid)
{
    // A tile: 64 rows x 64 bf16 = 512 x 16B chunks, 2 per thread
    #pragma unroll
    for (int i = 0; i < 2; ++i) {
        int idx = tid + i * 256;
        int r = idx >> 3, ch = idx & 7;
        async_b128(ldsA + r * 144 + ch * 16,
                   A + (size_t)(m0 + r) * lda + k0 + ch * 8);
    }
    // B tile: 256 rows x 64 bf16 = 2048 x 16B chunks, 8 per thread
    #pragma unroll
    for (int i = 0; i < 8; ++i) {
        int idx = tid + i * 256;
        int r = idx >> 3, ch = idx & 7;
        async_b128(ldsB + r * 144 + ch * 16,
                   Wt + (size_t)(n0 + r) * K + k0 + ch * 8);
    }
}

template <int EPI>
__global__ __launch_bounds__(256)
void gemm_bf16(const unsigned short* __restrict__ A, int lda,
               const unsigned short* __restrict__ Wt,
               const float* __restrict__ bias,
               const float* __restrict__ resid,
               const float* __restrict__ pos,
               float* __restrict__ outF,
               unsigned short* __restrict__ outH,
               int M, int N, int K, int ldout)
{
    __shared__ __align__(16) unsigned int lds_a[2][64 * 36];    // 2 x 9.2KB
    __shared__ __align__(16) unsigned int lds_b[2][256 * 36];   // 2 x 36.9KB

    const int tid  = threadIdx.x;
    const int m0   = blockIdx.y * 64;
    const int n0   = blockIdx.x * 256;
    const int wave = tid >> 5, lane = tid & 31;
    const int wm   = wave >> 2;      // 0..1 -> 32 rows
    const int wn   = wave & 3;       // 0..3 -> 64 cols
    const int lrow = lane & 15;
    const int hsel = lane >> 4;

    const vf8_t vzero = {0.f,0.f,0.f,0.f,0.f,0.f,0.f,0.f};
    vf8_t acc[2][4];
    #pragma unroll
    for (int i = 0; i < 2; ++i)
        #pragma unroll
        for (int j = 0; j < 4; ++j) acc[i][j] = vzero;

    const unsigned ldsA0 = (unsigned)(size_t)(void*)&lds_a[0][0];
    const unsigned ldsB0 = (unsigned)(size_t)(void*)&lds_b[0][0];
    const unsigned ABUF  = 64 * 36 * 4;
    const unsigned BBUF  = 256 * 36 * 4;

    const int nst = K >> 6;
    // prologue: stage 0 into buffer 0
    stage_copy(ldsA0, ldsB0, A, lda, Wt, K, m0, n0, 0, tid);

    for (int st = 0; st < nst; ++st) {
        const int buf = st & 1;
        wait_async0();        // this wave's stage-st copies landed
        __syncthreads();      // all waves' copies landed; stage st-1 reads done
        if (st + 1 < nst)     // prefetch next stage into the other buffer
            stage_copy(ldsA0 + (buf ^ 1) * ABUF, ldsB0 + (buf ^ 1) * BBUF,
                       A, lda, Wt, K, m0, n0, (st + 1) * 64, tid);

        const unsigned int* la = lds_a[buf];
        const unsigned int* lb = lds_b[buf];
        #pragma unroll
        for (int ks = 0; ks < 2; ++ks) {
            // ISA 7.12.2 16-bit A 16x32 layout: per-lane K pairs
            FragBF fa[2], fb[4];
            #pragma unroll
            for (int j = 0; j < 8; ++j) {
                int kw = (ks * 32 + ((j >> 2) << 4) + (hsel << 3) + ((j & 3) << 1)) >> 1;
                fa[0].u[j] = la[(wm * 32 +      lrow) * 36 + kw];
                fa[1].u[j] = la[(wm * 32 + 16 + lrow) * 36 + kw];
                fb[0].u[j] = lb[(wn * 64 +      lrow) * 36 + kw];
                fb[1].u[j] = lb[(wn * 64 + 16 + lrow) * 36 + kw];
                fb[2].u[j] = lb[(wn * 64 + 32 + lrow) * 36 + kw];
                fb[3].u[j] = lb[(wn * 64 + 48 + lrow) * 36 + kw];
            }
            #pragma unroll
            for (int si = 0; si < 2; ++si)
                #pragma unroll
                for (int sj = 0; sj < 4; ++sj)
                    acc[si][sj] = __builtin_amdgcn_wmma_f32_16x16x32_bf16(
                        false, fa[si].v, false, fb[sj].v,
                        (short)0, acc[si][sj], false, false);
        }
    }

    // ---- epilogue: C/D layout m = 8*hsel + v, n = lane%16 per 16x16 tile ----
    #pragma unroll
    for (int si = 0; si < 2; ++si) {
        #pragma unroll
        for (int sj = 0; sj < 4; ++sj) {
            #pragma unroll
            for (int v = 0; v < 8; ++v) {
                float a = acc[si][sj][v];
                int m = m0 + wm * 32 + si * 16 + hsel * 8 + v;
                int n = n0 + wn * 64 + sj * 16 + lrow;
                if (n >= N) continue;
                float r = a + bias[n];
                if (EPI == 0) {
                    outF[(size_t)m * ldout + n] = r;
                } else if (EPI == 1) {
                    int b = m / NPATCH, tok = m % NPATCH + 1;
                    outF[((size_t)(b * S1 + tok)) * DMODEL + n] =
                        r + pos[(size_t)tok * DMODEL + n];
                } else if (EPI == 2) {
                    outF[(size_t)m * ldout + n] = resid[(size_t)m * ldout + n] + r;
                } else {
                    float g = r * 0.5f * (1.0f + erff(r * 0.70710678118654752f));
                    outH[(size_t)m * ldout + n] = f32_to_bf16(g);
                }
            }
        }
    }
}

// =====================================================================
// LayerNorm over 768-elem rows -> bf16 output (feeds WMMA A operand)
// =====================================================================
__global__ __launch_bounds__(256)
void ln_kernel(const float* __restrict__ in, size_t row_stride,
               const float* __restrict__ g, const float* __restrict__ bb,
               unsigned short* __restrict__ out)
{
    __shared__ float red[256];
    const int row = blockIdx.x, tid = threadIdx.x;
    const float* p = in + (size_t)row * row_stride;
    float x0 = p[tid], x1 = p[tid + 256], x2 = p[tid + 512];
    red[tid] = x0 + x1 + x2;
    __syncthreads();
    for (int st = 128; st > 0; st >>= 1) { if (tid < st) red[tid] += red[tid + st]; __syncthreads(); }
    float mean = red[0] * (1.0f / 768.0f);
    __syncthreads();
    float d0 = x0 - mean, d1 = x1 - mean, d2 = x2 - mean;
    red[tid] = d0 * d0 + d1 * d1 + d2 * d2;
    __syncthreads();
    for (int st = 128; st > 0; st >>= 1) { if (tid < st) red[tid] += red[tid + st]; __syncthreads(); }
    float rs = rsqrtf(red[0] * (1.0f / 768.0f) + 1e-6f);
    size_t o = (size_t)row * DMODEL;
    out[o + tid]       = f32_to_bf16(d0 * rs * g[tid]       + bb[tid]);
    out[o + tid + 256] = f32_to_bf16(d1 * rs * g[tid + 256] + bb[tid + 256]);
    out[o + tid + 512] = f32_to_bf16(d2 * rs * g[tid + 512] + bb[tid + 512]);
}

// =====================================================================
// Attention: one block per (b, head); one wave per query. S in {197, 99}.
// qkv layout [B*S, 2304]: q|k|v at col offsets 0|768|1536, head h at +64h.
// ~3.4% of total FLOPs -> VALU f32 path.
// =====================================================================
__global__ __launch_bounds__(256)
void attn_kernel(const float* __restrict__ qkv, unsigned short* __restrict__ o, int S)
{
    __shared__ float sc[8][208];
    const int bh = blockIdx.x;
    const int b = bh / H_HEADS, h = bh % H_HEADS;
    const int wave = threadIdx.x >> 5, lane = threadIdx.x & 31;
    const float scale = 0.125f;   // 1/sqrt(64)

    for (int q = wave; q < S; q += 8) {
        const float* qp = qkv + ((size_t)(b * S + q)) * 2304 + h * HDIM;
        float q0 = qp[lane * 2], q1 = qp[lane * 2 + 1];
        for (int kk = 0; kk < S; ++kk) {
            const float* kp = qkv + ((size_t)(b * S + kk)) * 2304 + 768 + h * HDIM;
            float d = q0 * kp[lane * 2] + q1 * kp[lane * 2 + 1];
            for (int off = 16; off > 0; off >>= 1) d += __shfl_xor(d, off, 32);
            if (lane == 0) sc[wave][kk] = d * scale;
        }
        __threadfence_block();
        float mx = -1e30f;
        for (int kk = lane; kk < S; kk += 32) mx = fmaxf(mx, sc[wave][kk]);
        for (int off = 16; off > 0; off >>= 1) mx = fmaxf(mx, __shfl_xor(mx, off, 32));
        float sum = 0.0f;
        for (int kk = lane; kk < S; kk += 32) {
            float e = __expf(sc[wave][kk] - mx);
            sc[wave][kk] = e; sum += e;
        }
        for (int off = 16; off > 0; off >>= 1) sum += __shfl_xor(sum, off, 32);
        float inv = 1.0f / sum;
        __threadfence_block();
        float a0 = 0.0f, a1 = 0.0f;
        for (int kk = 0; kk < S; ++kk) {
            const float* vp = qkv + ((size_t)(b * S + kk)) * 2304 + 1536 + h * HDIM;
            float p = sc[wave][kk] * inv;
            a0 += p * vp[lane * 2]; a1 += p * vp[lane * 2 + 1];
        }
        unsigned short* op = o + ((size_t)(b * S + q)) * DMODEL + h * HDIM;
        op[lane * 2]     = f32_to_bf16(a0);
        op[lane * 2 + 1] = f32_to_bf16(a1);
    }
}

// =====================================================================
// Patch extraction (im2col) -> bf16 rows [B*196, 768]
// =====================================================================
__global__ __launch_bounds__(256)
void im2col_kernel(const float* __restrict__ x, unsigned short* __restrict__ patches)
{
    const int m = blockIdx.x;                  // b*196 + n
    const int b = m / NPATCH, n = m % NPATCH;
    const int gy = n / 14, gx = n % 14;
    const int tid = threadIdx.x;
    for (int i = tid; i < 768; i += 256) {
        int c = i >> 8, rem = i & 255, py = rem >> 4, px = rem & 15;
        float v = x[(((size_t)b * 3 + c) * 224 + (gy * 16 + py)) * 224 + (gx * 16 + px)];
        patches[(size_t)m * 768 + i] = f32_to_bf16(v);
    }
}

__global__ __launch_bounds__(256)
void cls_init_kernel(const float* __restrict__ cls, const float* __restrict__ pos,
                     float* __restrict__ t)
{
    const int b = blockIdx.x, tid = threadIdx.x;
    float* d = t + ((size_t)b * S1) * DMODEL;
    for (int i = tid; i < DMODEL; i += 256) d[i] = cls[i] + pos[i];
}

// =====================================================================
// Pruning: L2 norms of patch tokens, per-batch top-98, gather -> t2[B,99,768]
// =====================================================================
__global__ __launch_bounds__(256)
void norm_kernel(const float* __restrict__ t, float* __restrict__ norms)
{
    __shared__ float red[256];
    const int m = blockIdx.x;                  // b*196 + n
    const int b = m / NPATCH, n = m % NPATCH;
    const float* p = t + ((size_t)(b * S1 + 1 + n)) * DMODEL;
    const int tid = threadIdx.x;
    float x0 = p[tid], x1 = p[tid + 256], x2 = p[tid + 512];
    red[tid] = x0 * x0 + x1 * x1 + x2 * x2;
    __syncthreads();
    for (int st = 128; st > 0; st >>= 1) { if (tid < st) red[tid] += red[tid + st]; __syncthreads(); }
    if (tid == 0) norms[m] = sqrtf(red[0]);
}

__global__ __launch_bounds__(256)
void topk_kernel(const float* __restrict__ norms, int* __restrict__ idx)
{
    __shared__ float val[256];
    __shared__ int   vi[256];
    __shared__ float cand[NPATCH];
    const int b = blockIdx.x, tid = threadIdx.x;
    for (int i = tid; i < NPATCH; i += 256) cand[i] = norms[b * NPATCH + i];
    __syncthreads();
    for (int j = 0; j < KEEPN; ++j) {
        val[tid] = (tid < NPATCH) ? cand[tid] : -1e30f;
        vi[tid]  = tid;
        __syncthreads();
        for (int st = 128; st > 0; st >>= 1) {
            if (tid < st) {
                float v2 = val[tid + st]; int i2 = vi[tid + st];
                if (v2 > val[tid] || (v2 == val[tid] && i2 < vi[tid])) { val[tid] = v2; vi[tid] = i2; }
            }
            __syncthreads();
        }
        if (tid == 0) { idx[b * KEEPN + j] = vi[0]; cand[vi[0]] = -1e30f; }
        __syncthreads();
    }
}

__global__ __launch_bounds__(256)
void gather_kernel(const float* __restrict__ t, const int* __restrict__ idx,
                   float* __restrict__ t2)
{
    const int m = blockIdx.x;                  // b*99 + r
    const int b = m / S2, r = m % S2;
    const int src = (r == 0) ? 0 : (1 + idx[b * KEEPN + (r - 1)]);
    const float* s = t + ((size_t)(b * S1 + src)) * DMODEL;
    float* d = t2 + (size_t)m * DMODEL;
    const int tid = threadIdx.x;
    d[tid] = s[tid]; d[tid + 256] = s[tid + 256]; d[tid + 512] = s[tid + 512];
}

// =====================================================================
// Host launcher
// =====================================================================
extern "C" void kernel_launch(void* const* d_in, const int* in_sizes, int n_in,
                              void* d_out, int out_size, void* d_ws, size_t ws_size,
                              hipStream_t stream)
{
    (void)in_sizes; (void)n_in; (void)out_size; (void)ws_size;
    const float* x        = (const float*)d_in[0];
    const float* patch_w  = (const float*)d_in[1];
    const float* patch_b  = (const float*)d_in[2];
    const float* cls_tok  = (const float*)d_in[3];
    const float* pos_emb  = (const float*)d_in[4];
    const float* ln1_g    = (const float*)d_in[5];
    const float* ln1_b    = (const float*)d_in[6];
    const float* qkv_w    = (const float*)d_in[7];
    const float* qkv_b    = (const float*)d_in[8];
    const float* proj_w   = (const float*)d_in[9];
    const float* proj_b   = (const float*)d_in[10];
    const float* ln2_g    = (const float*)d_in[11];
    const float* ln2_b    = (const float*)d_in[12];
    const float* fc1_w    = (const float*)d_in[13];
    const float* fc1_b    = (const float*)d_in[14];
    const float* fc2_w    = (const float*)d_in[15];
    const float* fc2_b    = (const float*)d_in[16];
    const float* norm_g   = (const float*)d_in[17];
    const float* norm_b   = (const float*)d_in[18];
    const float* head_w   = (const float*)d_in[19];
    const float* head_b   = (const float*)d_in[20];

    char* ws = (char*)d_ws;
    size_t off = 0;
    auto alloc = [&](size_t bytes) -> void* {
        void* p = ws + off;
        off += (bytes + 255) & ~(size_t)255;
        return p;
    };
    float*          t       = (float*)         alloc((size_t)BATCH * S1 * DMODEL * 4);
    float*          t2      = (float*)         alloc((size_t)BATCH * S2 * DMODEL * 4);
    unsigned short* hbuf    = (unsigned short*)alloc((size_t)BATCH * S1 * DMODEL * 2);
    float*          qkvbuf  = (float*)         alloc((size_t)BATCH * S1 * 3 * DMODEL * 4);
    unsigned short* obuf    = (unsigned short*)alloc((size_t)BATCH * S1 * DMODEL * 2);
    unsigned short* ubuf    = (unsigned short*)alloc((size_t)BATCH * S1 * FFDIM * 2);
    unsigned short* patches = (unsigned short*)alloc((size_t)BATCH * NPATCH * DMODEL * 2);
    float*          norms   = (float*)         alloc((size_t)BATCH * NPATCH * 4);
    int*            idxb    = (int*)           alloc((size_t)BATCH * KEEPN * 4);
    unsigned short* clsln   = (unsigned short*)alloc((size_t)BATCH * DMODEL * 2);
    // pre-transposed bf16 weights [N, K]
    unsigned short* wt_patch = (unsigned short*)alloc((size_t)DMODEL * DMODEL * 2);
    unsigned short* wt_qkv   = (unsigned short*)alloc((size_t)NLAYER * 3 * DMODEL * DMODEL * 2);
    unsigned short* wt_proj  = (unsigned short*)alloc((size_t)NLAYER * DMODEL * DMODEL * 2);
    unsigned short* wt_fc1   = (unsigned short*)alloc((size_t)NLAYER * FFDIM * DMODEL * 2);
    unsigned short* wt_fc2   = (unsigned short*)alloc((size_t)NLAYER * DMODEL * FFDIM * 2);
    unsigned short* wt_head  = (unsigned short*)alloc((size_t)NCLS_P * DMODEL * 2);

    // ---- one-shot weight convert + transpose (bandwidth-bound) ----
    auto conv = [&](const float* W, unsigned short* Wt, int K, int N, int Np) {
        wconv_kernel<<<dim3(Np / 32, K / 32), 256, 0, stream>>>(W, Wt, K, N);
    };
    conv(patch_w, wt_patch, DMODEL, DMODEL, DMODEL);
    conv(head_w,  wt_head,  DMODEL, NCLS,   NCLS_P);
    for (int i = 0; i < NLAYER; ++i) {
        conv(qkv_w  + (size_t)i * DMODEL * 3 * DMODEL, wt_qkv  + (size_t)i * 3 * DMODEL * DMODEL, DMODEL, 3 * DMODEL, 3 * DMODEL);
        conv(proj_w + (size_t)i * DMODEL * DMODEL,     wt_proj + (size_t)i * DMODEL * DMODEL,     DMODEL, DMODEL,     DMODEL);
        conv(fc1_w  + (size_t)i * DMODEL * FFDIM,      wt_fc1  + (size_t)i * FFDIM * DMODEL,      DMODEL, FFDIM,      FFDIM);
        conv(fc2_w  + (size_t)i * FFDIM * DMODEL,      wt_fc2  + (size_t)i * DMODEL * FFDIM,      FFDIM,  DMODEL,     DMODEL);
    }

    // ---- patch embed + cls + positional embedding ----
    im2col_kernel<<<BATCH * NPATCH, 256, 0, stream>>>(x, patches);
    cls_init_kernel<<<BATCH, 256, 0, stream>>>(cls_tok, pos_emb, t);
    gemm_bf16<1><<<dim3(3, (BATCH * NPATCH) / 64), 256, 0, stream>>>(
        patches, DMODEL, wt_patch, patch_b, nullptr, pos_emb,
        t, nullptr, BATCH * NPATCH, DMODEL, DMODEL, DMODEL);

    auto layer = [&](float* tt, int S, int i) {
        const int M = BATCH * S;
        ln_kernel<<<M, 256, 0, stream>>>(tt, DMODEL, ln1_g + (size_t)i * DMODEL,
                                         ln1_b + (size_t)i * DMODEL, hbuf);
        gemm_bf16<0><<<dim3(9, M / 64), 256, 0, stream>>>(
            hbuf, DMODEL, wt_qkv + (size_t)i * 3 * DMODEL * DMODEL,
            qkv_b + (size_t)i * 3 * DMODEL, nullptr, nullptr,
            qkvbuf, nullptr, M, 3 * DMODEL, DMODEL, 3 * DMODEL);
        attn_kernel<<<BATCH * H_HEADS, 256, 0, stream>>>(qkvbuf, obuf, S);
        gemm_bf16<2><<<dim3(3, M / 64), 256, 0, stream>>>(
            obuf, DMODEL, wt_proj + (size_t)i * DMODEL * DMODEL,
            proj_b + (size_t)i * DMODEL, tt, nullptr,
            tt, nullptr, M, DMODEL, DMODEL, DMODEL);
        ln_kernel<<<M, 256, 0, stream>>>(tt, DMODEL, ln2_g + (size_t)i * DMODEL,
                                         ln2_b + (size_t)i * DMODEL, hbuf);
        gemm_bf16<3><<<dim3(12, M / 64), 256, 0, stream>>>(
            hbuf, DMODEL, wt_fc1 + (size_t)i * FFDIM * DMODEL,
            fc1_b + (size_t)i * FFDIM, nullptr, nullptr,
            nullptr, ubuf, M, FFDIM, DMODEL, FFDIM);
        gemm_bf16<2><<<dim3(3, M / 64), 256, 0, stream>>>(
            ubuf, FFDIM, wt_fc2 + (size_t)i * DMODEL * FFDIM,
            fc2_b + (size_t)i * DMODEL, tt, nullptr,
            tt, nullptr, M, DMODEL, FFDIM, DMODEL);
    };

    for (int i = 0; i < PRUNE_L; ++i) layer(t, S1, i);

    // ---- dynamic token pruning ----
    norm_kernel<<<BATCH * NPATCH, 256, 0, stream>>>(t, norms);
    topk_kernel<<<BATCH, 256, 0, stream>>>(norms, idxb);
    gather_kernel<<<BATCH * S2, 256, 0, stream>>>(t, idxb, t2);

    for (int i = PRUNE_L; i < NLAYER; ++i) layer(t2, S2, i);

    // ---- final LN (cls rows only) + classifier head ----
    ln_kernel<<<BATCH, 256, 0, stream>>>(t2, (size_t)S2 * DMODEL, norm_g, norm_b, clsln);
    gemm_bf16<0><<<dim3(4, 1), 256, 0, stream>>>(
        clsln, DMODEL, wt_head, head_b, nullptr, nullptr,
        (float*)d_out, nullptr, BATCH, NCLS, DMODEL, NCLS);
}
